// Network_4973572129150
// MI455X (gfx1250) — matrix operand order repaired and checked by reading, not compiled, not run on hardware
//
#include <hip/hip_runtime.h>
#include <math.h>

// ---------------- problem constants (from the reference) ----------------
#define NNODES   6144
#define NEDGES   67584
#define NATOM    12
#define EPM      132          // edges per molecule (12*11)
#define BMOL     512
#define SWISH_C  1.6768f
#define TANH_C   1.5927f
#define SIG_C    1.8467f

typedef __attribute__((ext_vector_type(16))) _Float16 v16h;
typedef __attribute__((ext_vector_type(8)))  float    v8f;

// =======================================================================
// Generic WMMA GEMM:  Y[m, n] (+)= scale * sum_k A[m,k] * W[k,n]
//   A row m maps to (node = m/NCOMP, comp = m%NCOMP):
//       A[m,k] = X[node*ldx + xoff + k*NCOMP + comp]
//   Y scatter:  Y[node*ldy + yoff + n*NCOMP + comp]
//   W row-major (K x Nout); K and Nout zero-padded in the loaders.
//
// One wave per 32x32 output tile: 2x2 sub-tiles -> 4 v_wmma per K-step.
// f32 tiles are DMA'd into LDS with GLOBAL_LOAD_ASYNC_TO_LDS_B32
// (ASYNCcnt), OOB slots zero-filled with DS stores, then fragments are
// built with an f32->f16 convert and fed to v_wmma_f32_16x16x32_f16.
// M is always a multiple of 32 for every call in this network.
// =======================================================================
template<int NCOMP>
__global__ __launch_bounds__(32)
void gemm_wmma(const float* __restrict__ X, int ldx, int xoff,
               const float* __restrict__ W, int K, int Nout,
               float* __restrict__ Y, int ldy, int yoff,
               int M, float scale, int acc, int act)
{
    __shared__ float Asf[32 * 32];   // 32 rows x 32 k
    __shared__ float Bsf[32 * 32];   // 32 k    x 32 n
    const int t  = threadIdx.x;      // 0..31
    const int m0 = blockIdx.x * 32;
    const int n0 = blockIdx.y * 32;

    v8f c00 = {}, c01 = {}, c10 = {}, c11 = {};

    for (int k0 = 0; k0 < K; k0 += 32) {
        // WAR guard: LDS reads of the previous K-step must retire before the
        // async DMA engine overwrites the tiles (DS and async paths are not
        // mutually ordered in hardware).
        asm volatile("s_wait_dscnt 0x0" ::: "memory");

        const int  kk  = k0 + t;
        const bool kin = (kk < K);

        // ---- A tile: lane t owns k-column kk for rows m0..m0+31 ----
        if (kin) {
            #pragma unroll 1
            for (int i = 0; i < 32; ++i) {
                const int m    = m0 + i;
                const int node = m / NCOMP;
                const int comp = m - node * NCOMP;
                const float* ga = X + (long)node * ldx + xoff + kk * NCOMP + comp;
                const unsigned ldsa = (unsigned)(unsigned long long)&Asf[i * 32 + t];
                asm volatile("global_load_async_to_lds_b32 %0, %1, off"
                             :: "v"(ldsa), "v"((unsigned long long)ga) : "memory");
            }
        } else {
            #pragma unroll 1
            for (int i = 0; i < 32; ++i) Asf[i * 32 + t] = 0.0f;
        }

        // ---- B tile: lane t owns k-row kk for cols n0..n0+31 ----
        #pragma unroll 1
        for (int n = 0; n < 32; ++n) {
            if (kin && (n0 + n) < Nout) {
                const float* gw = W + (long)kk * Nout + n0 + n;
                const unsigned ldsb = (unsigned)(unsigned long long)&Bsf[t * 32 + n];
                asm volatile("global_load_async_to_lds_b32 %0, %1, off"
                             :: "v"(ldsb), "v"((unsigned long long)gw) : "memory");
            } else {
                Bsf[t * 32 + n] = 0.0f;
            }
        }

        // DMA completion before any fragment read
        asm volatile("s_wait_asynccnt 0x0" ::: "memory");

        // ---- build f16 fragments per CDNA5 16x16x32 layouts ----
        const int row  = t & 15;
        const int half = t >> 4;
        const int kb   = half * 16;
        v16h a0, a1, b0, b1;
        {
            const float* r0 = &Asf[row * 32];
            const float* r1 = &Asf[(row + 16) * 32];
            for (int e = 0; e < 8; ++e) {
                a0[e]     = (_Float16)r0[half * 8 + e];        // K = half*8 + e
                a0[e + 8] = (_Float16)r0[16 + half * 8 + e];   // K = 16 + half*8 + e
                a1[e]     = (_Float16)r1[half * 8 + e];
                a1[e + 8] = (_Float16)r1[16 + half * 8 + e];
            }
            for (int e = 0; e < 16; ++e) {
                b0[e] = (_Float16)Bsf[(half * 8 + e + (e & 8)) * 32 + row];        // col = lane&15
                b1[e] = (_Float16)Bsf[(half * 8 + e + (e & 8)) * 32 + 16 + row];   // col+16
            }
        }
        c00 = __builtin_amdgcn_wmma_f32_16x16x32_f16(false, a0, false, b0, (short)0, c00, false, false);
        c01 = __builtin_amdgcn_wmma_f32_16x16x32_f16(false, a0, false, b1, (short)0, c01, false, false);
        c10 = __builtin_amdgcn_wmma_f32_16x16x32_f16(false, a1, false, b0, (short)0, c10, false, false);
        c11 = __builtin_amdgcn_wmma_f32_16x16x32_f16(false, a1, false, b1, (short)0, c11, false, false);
    }

    // ---- epilogue: scale, optional accumulate, optional swish ----
    v8f ct[2][2] = { { c00, c01 }, { c10, c11 } };
    const int nn   = t & 15;
    const int half = t >> 4;
    for (int rt = 0; rt < 2; ++rt) {
        for (int cb = 0; cb < 2; ++cb) {
            const int col = n0 + cb * 16 + nn;
            if (col >= Nout) continue;
            for (int r = 0; r < 8; ++r) {
                const int m    = m0 + rt * 16 + half * 8 + r;
                const int node = m / NCOMP;
                const int comp = m - node * NCOMP;
                const long idx = (long)node * ldy + yoff + col * NCOMP + comp;
                float v = ct[rt][cb][r] * scale;
                if (acc) v += Y[idx];
                if (act) v = SWISH_C * v / (1.0f + __expf(-v));   // SWISH_C*swish
                Y[idx] = v;
            }
        }
    }
}

// =======================================================================
// Edge geometry: esh (E x 4), Gaussian basis gb (E x 64, 50 valid), ez.
// =======================================================================
__global__ void edge_geom_kernel(const float* __restrict__ pos,
                                 const int* __restrict__ esrc,
                                 const int* __restrict__ edst,
                                 const int* __restrict__ zatom,
                                 float* __restrict__ esh,
                                 float* __restrict__ gb,
                                 int* __restrict__ ez)
{
    const int e = blockIdx.x * 256 + threadIdx.x;
    if (e >= NEDGES) return;
    const int s = esrc[e], d = edst[e];
    const float dx = pos[3 * s + 0] - pos[3 * d + 0];
    const float dy = pos[3 * s + 1] - pos[3 * d + 1];
    const float dz = pos[3 * s + 2] - pos[3 * d + 2];
    const float el  = sqrtf(dx * dx + dy * dy + dz * dz);
    const float inv = 1.0f / el;
    const float SQ3 = 1.7320508075688772f;
    const float ec  = 0.5f * (__cosf(0.3141592653589793f * el) + 1.0f); // cos(pi*el/10)
    const float sc  = ec * 0.22360679774997896f;                        // /sqrt(20)
    esh[4 * e + 0] = sc;
    esh[4 * e + 1] = sc * SQ3 * dx * inv;
    esh[4 * e + 2] = sc * SQ3 * dy * inv;
    esh[4 * e + 3] = sc * SQ3 * dz * inv;
    const float tt = el * 4.9f;                   // el/sig, sig = 10/49, mu_i = i*sig
    for (int i = 0; i < 64; ++i) {
        float v = 0.0f;
        if (i < 50) { const float dd = tt - (float)i; v = __expf(-0.5f * dd * dd); }
        gb[64L * e + i] = v;
    }
    const int tab[10] = { -1, 0, -1, -1, -1, -1, 1, 2, 3, 4 };
    ez[e] = 5 * tab[zatom[s]] + tab[zatom[d]];
}

// =======================================================================
// Attention: softmax over 8 filters of (concat(z_emb[ez], el_emb) . key)/16
// =======================================================================
__global__ void attention_kernel(const float* __restrict__ elemb,
                                 const float* __restrict__ zemb,
                                 const int* __restrict__ ez,
                                 const float* __restrict__ keys,
                                 float* __restrict__ att)
{
    const int e = blockIdx.x * 256 + threadIdx.x;
    if (e >= NEDGES) return;
    const float* ze = zemb + 128L * ez[e];
    const float* em = elemb + 128L * e;
    float lg[8];
    float mx = -1e30f;
    for (int f = 0; f < 8; ++f) {
        const float* kf = keys + 256L * f;
        float accv = 0.0f;
        for (int c = 0; c < 128; ++c) accv += ze[c] * kf[c];
        for (int c = 0; c < 128; ++c) accv += em[c] * kf[128 + c];
        lg[f] = accv * 0.0625f;                   // / sqrt(256)
        mx = fmaxf(mx, lg[f]);
    }
    float ssum = 0.0f;
    for (int f = 0; f < 8; ++f) { lg[f] = __expf(lg[f] - mx); ssum += lg[f]; }
    const float invs = 1.0f / ssum;
    for (int f = 0; f < 8; ++f) att[8L * e + f] = lg[f] * invs;
}

// =======================================================================
// Initial node features: h = segment_sum(esh, edge_src); h[:,0] = 1.
// Outgoing edges of node (mol, si) are the contiguous block si*11..si*11+10.
// =======================================================================
__global__ void hinit_kernel(const float* __restrict__ esh, float* __restrict__ h)
{
    const int n = blockIdx.x * 256 + threadIdx.x;
    if (n >= NNODES) return;
    const int mol = n / NATOM, si = n - mol * NATOM;
    float a1 = 0.f, a2 = 0.f, a3 = 0.f;
    for (int j = 0; j < 11; ++j) {
        const long e = (long)mol * EPM + si * 11 + j;
        a1 += esh[4 * e + 1]; a2 += esh[4 * e + 2]; a3 += esh[4 * e + 3];
    }
    h[4L * n + 0] = 1.0f; h[4L * n + 1] = a1; h[4L * n + 2] = a2; h[4L * n + 3] = a3;
}

// =======================================================================
// Tensor product + segment-sum, one workgroup per DESTINATION node.
// Kinds: 0 s*s->s, 1 s*v->v, 2 v*s->v, 3 v.v->s, 4 v x v->v.
// Per-layer static instruction tables; `a` includes CG and path norms.
// Output element (ooff+idx) is owned by thread idx%256 -> race free.
// =======================================================================
struct TpInstr { short xoff, ooff, m1, woff, kind; float a; };

__constant__ TpInstr TP_L0[5] = {
    {0, 0, 1, 0, 0, 0.70710678f},
    {0, 1, 1, 1, 1, 0.70710678f},
    {1, 1, 1, 2, 2, 0.70710678f},
    {1, 0, 1, 3, 3, 0.40824829f},
    {1, 4, 1, 4, 4, 0.70710678f},
};
__constant__ TpInstr TP_L1[8] = {
    {0,     0,  256, 0,   0, 1.0f},
    {0,   256,  256, 256, 1, 1.0f},
    {256, 1024, 16,  512, 2, 0.70710678f},
    {256, 1072, 16,  528, 3, 0.57735027f},
    {256, 1088, 16,  544, 4, 0.5f},
    {304, 1088, 16,  560, 2, 0.70710678f},
    {304, 1136, 16,  576, 3, 0.57735027f},
    {304, 1024, 16,  592, 4, 0.5f},
};
__constant__ TpInstr TP_L2[10] = {
    {0,     0,  256, 0,    0, 1.0f},
    {0,   256,  256, 256,  1, 1.0f},
    {256, 1024, 256, 512,  0, 1.0f},
    {256, 1280, 256, 768,  1, 1.0f},
    {512, 2048, 16,  1024, 2, 0.70710678f},
    {512, 2096, 16,  1040, 3, 0.57735027f},
    {512, 2112, 16,  1056, 4, 0.5f},
    {560, 2112, 16,  1072, 2, 0.70710678f},
    {560, 2160, 16,  1088, 3, 0.57735027f},
    {560, 2048, 16,  1104, 4, 0.5f},
};
__constant__ TpInstr TP_L3[4] = {
    {0,    0,  256, 0,   0, 1.0f},
    {256, 256, 256, 256, 0, 1.0f},
    {512, 512, 16,  512, 3, 0.57735027f},
    {560, 528, 16,  528, 3, 0.57735027f},
};

__global__ __launch_bounds__(256)
void tp_aggregate(const float* __restrict__ xx, int ldx,
                  const float* __restrict__ esh,
                  const float* __restrict__ att,
                  const float* __restrict__ tpw, int numel,
                  float* __restrict__ agg, int tpdim, int layer)
{
    __shared__ float sagg[2176];
    const int n = blockIdx.x, t = threadIdx.x;
    for (int j = t; j < tpdim; j += 256) sagg[j] = 0.0f;
    __syncthreads();

    const TpInstr* tab; int ni;
    switch (layer) {
        case 0:  tab = TP_L0; ni = 5;  break;
        case 1:  tab = TP_L1; ni = 8;  break;
        case 2:  tab = TP_L2; ni = 10; break;
        default: tab = TP_L3; ni = 4;  break;
    }

    const int mol = n / NATOM, di = n - mol * NATOM;
    for (int s = 0; s < NATOM; ++s) {
        if (s == di) continue;
        const int  k   = s * 11 + (di < s ? di : di - 1);
        const long e   = (long)mol * EPM + k;
        const int  src = mol * NATOM + s;
        const float y0 = esh[4 * e + 0], y1 = esh[4 * e + 1];
        const float y2 = esh[4 * e + 2], y3 = esh[4 * e + 3];
        float af[8];
        for (int f = 0; f < 8; ++f) af[f] = att[8 * e + f];
        const float* xp = xx + (long)src * ldx;

        for (int ii = 0; ii < ni; ++ii) {
            const TpInstr I = tab[ii];
            const int odim  = (I.kind == 0 || I.kind == 3) ? 1 : 3;
            const int total = I.m1 * odim;
            for (int idx = t; idx < total; idx += 256) {
                const int u  = (odim == 1) ? idx : idx / 3;
                const int cc = (odim == 1) ? 0   : idx - u * 3;
                float w = 0.0f;
                for (int f = 0; f < 8; ++f) w += af[f] * tpw[(long)f * numel + I.woff + u];
                float v;
                switch (I.kind) {
                    case 0: v = xp[I.xoff + u] * y0; break;
                    case 1: v = xp[I.xoff + u] * ((cc == 0) ? y1 : (cc == 1) ? y2 : y3); break;
                    case 2: v = xp[I.xoff + u * 3 + cc] * y0; break;
                    case 3: v = xp[I.xoff + u * 3 + 0] * y1 +
                                xp[I.xoff + u * 3 + 1] * y2 +
                                xp[I.xoff + u * 3 + 2] * y3; break;
                    default: {
                        const float xa = xp[I.xoff + u * 3 + 0];
                        const float xb = xp[I.xoff + u * 3 + 1];
                        const float xc = xp[I.xoff + u * 3 + 2];
                        v = (cc == 0) ? (xb * y3 - xc * y2)
                          : (cc == 1) ? (xc * y1 - xa * y3)
                                      : (xa * y2 - xb * y1);
                    }
                }
                sagg[I.ooff + idx] += I.a * w * v;
            }
        }
    }
    __syncthreads();
    for (int j = t; j < tpdim; j += 256) agg[(long)n * tpdim + j] = sagg[j];
}

// =======================================================================
// Gate: [swish 256][tanh ntanh][32 sigmoid gates][96 gated vector comps]
// =======================================================================
__global__ void gate_kernel(const float* __restrict__ xin, int ldin,
                            float* __restrict__ hout, int ldout,
                            int ntanh, int total)
{
    const int i = blockIdx.x * 256 + threadIdx.x;
    if (i >= total) return;
    const int n = i / ldout, j = i - n * ldout;
    const float* xr = xin + (long)n * ldin;
    float v;
    if (j < 256) {
        const float x = xr[j];
        v = SWISH_C * x / (1.0f + __expf(-x));
    } else if (j < 256 + ntanh) {
        v = TANH_C * tanhf(xr[j]);
    } else {
        const int rel = j - 256 - ntanh;
        const int u   = rel / 3;
        float g = xr[256 + ntanh + u];
        g = SIG_C / (1.0f + __expf(-g));
        v = xr[256 + ntanh + 32 + rel] * g;
    }
    hout[(long)n * ldout + j] = v;
}

// =======================================================================
// Readout: out[mol] = sum_atoms (y0 + 0.5*y1^2) / sqrt(20)
// =======================================================================
__global__ void readout_kernel(const float* __restrict__ sout, float* __restrict__ out)
{
    const int m = blockIdx.x * 256 + threadIdx.x;
    if (m >= BMOL) return;
    float accv = 0.0f;
    for (int a = 0; a < NATOM; ++a) {
        const int n = m * NATOM + a;
        const float y0 = sout[2 * n + 0], y1 = sout[2 * n + 1];
        accv += y0 + 0.5f * y1 * y1;
    }
    out[m] = accv * 0.22360679774997896f;
}

// =======================================================================
// Host side
// =======================================================================
static inline void lin(hipStream_t st, const float* X, int ldx, int xoff, int ncomp,
                       const float* W, int K, int Nout,
                       float* Y, int ldy, int yoff, int M, float scale, int acc, int act)
{
    dim3 grid((M + 31) / 32, (Nout + 31) / 32);
    if (ncomp == 1)
        gemm_wmma<1><<<grid, dim3(32), 0, st>>>(X, ldx, xoff, W, K, Nout, Y, ldy, yoff, M, scale, acc, act);
    else
        gemm_wmma<3><<<grid, dim3(32), 0, st>>>(X, ldx, xoff, W, K, Nout, Y, ldy, yoff, M, scale, acc, act);
}

extern "C" void kernel_launch(void* const* d_in, const int* in_sizes, int n_in,
                              void* d_out, int out_size, void* d_ws, size_t ws_size,
                              hipStream_t stream)
{
    (void)in_sizes; (void)n_in; (void)out_size; (void)ws_size;
    const int N = NNODES, E = NEDGES;

    // ---- walk inputs in setup_inputs() / params insertion order ----
    int ci = 0;
    const float* pos   = (const float*)d_in[ci++];
    const float* zemb  = (const float*)d_in[ci++];   // params.z_emb (25,128)
    const float* radW1 = (const float*)d_in[ci++];   // (50,128)
    const float* radW2 = (const float*)d_in[ci++];   // (128,128)
    const float* SI[4][5]; const float* LN1[4][4]; const float* LN2[4][10];
    const float* KEYS[4]; const float* TPWt[4];
    const int nsi[4] = {2, 4, 5, 2}, nl1[4] = {2, 3, 4, 4}, nl2[4] = {3, 8, 10, 4};
    for (int L = 0; L < 4; ++L) {
        for (int i = 0; i < nsi[L]; ++i) SI[L][i]  = (const float*)d_in[ci++];
        for (int i = 0; i < nl1[L]; ++i) LN1[L][i] = (const float*)d_in[ci++];
        for (int i = 0; i < nl2[L]; ++i) LN2[L][i] = (const float*)d_in[ci++];
        KEYS[L] = (const float*)d_in[ci++];
        TPWt[L] = (const float*)d_in[ci++];
    }
    const int* zatom = (const int*)d_in[ci++];
    const int* esrc  = (const int*)d_in[ci++];
    const int* edst  = (const int*)d_in[ci++];
    /* batch unused: node->molecule mapping is analytic (n / 12) */

    // ---- workspace layout ----
    float* p = (float*)d_ws;
    float* esh   = p; p += (size_t)E * 4;
    float* gb    = p; p += (size_t)E * 64;
    float* h1    = p; p += (size_t)E * 128;
    float* elemb = p; p += (size_t)E * 128;
    float* att   = p; p += (size_t)E * 8;
    int*   ez    = (int*)p; p += (size_t)E;
    float* hbuf  = p; p += (size_t)N * 640;
    float* xx    = p; p += (size_t)N * 640;
    float* sout  = p; p += (size_t)N * 640;
    float* agg   = p; p += (size_t)N * 2176;

    const float inv16 = 0.0625f;                 // 1/sqrt(256)
    const float q     = 0.25f;                   // 1/sqrt(16)
    const float sc272 = 0.1f / 16.492422502470642f;  // 0.1/sqrt(272)
    const float sc16g = 0.025f;                  // 0.1/sqrt(16)

    const dim3 eb((E + 255) / 256), tb(256);
    const dim3 nb((N + 255) / 256);

    // ---- edge geometry + radial MLP (WMMA) + initial features ----
    edge_geom_kernel<<<eb, tb, 0, stream>>>(pos, esrc, edst, zatom, esh, gb, ez);
    lin(stream, gb, 64, 0, 1, radW1, 50, 128, h1, 128, 0, E, 1.0f, 0, 1);
    lin(stream, h1, 128, 0, 1, radW2, 128, 128, elemb, 128, 0, E, 0.08838834764831845f, 0, 1);
    hinit_kernel<<<nb, tb, 0, stream>>>(esh, hbuf);

    // =================== layer 0: (N,4) -> (N,384) -> gate -> (N,352) ===================
    attention_kernel<<<eb, tb, 0, stream>>>(elemb, zemb, ez, KEYS[0], att);
    lin(stream, hbuf, 4, 0, 1, LN1[0][0], 1, 1, xx, 4, 0, N, 1.0f, 0, 0);
    lin(stream, hbuf, 4, 1, 3, LN1[0][1], 1, 1, xx, 4, 1, 3 * N, 1.0f, 0, 0);
    hipMemsetAsync(sout, 0, (size_t)N * 640 * 4, stream);
    lin(stream, hbuf, 4, 0, 1, SI[0][0], 1, 288, sout, 384, 0,   N,     1.0f, 0, 0);
    lin(stream, hbuf, 4, 1, 3, SI[0][1], 1, 16,  sout, 384, 288, 3 * N, 1.0f, 0, 0);
    tp_aggregate<<<N, tb, 0, stream>>>(xx, 4, esh, att, TPWt[0], 5, agg, 7, 0);
    lin(stream, agg, 7, 0, 1, LN2[0][0], 1, 288, sout, 384, 0,   N,     0.1f, 1, 0);
    lin(stream, agg, 7, 1, 3, LN2[0][1], 1, 16,  sout, 384, 288, 3 * N, 0.1f, 1, 0);
    lin(stream, agg, 7, 4, 3, LN2[0][2], 1, 16,  sout, 384, 336, 3 * N, 0.1f, 1, 0);
    gate_kernel<<<(N * 352 + 255) / 256, tb, 0, stream>>>(sout, 384, hbuf, 352, 0, N * 352);

    // =================== layer 1: (N,352) -> (N,640) -> gate -> (N,608) ===================
    attention_kernel<<<eb, tb, 0, stream>>>(elemb, zemb, ez, KEYS[1], att);
    lin(stream, hbuf, 352, 0,   1, LN1[1][0], 256, 256, xx, 352, 0,   N,     inv16, 0, 0);
    lin(stream, hbuf, 352, 256, 3, LN1[1][1], 16,  16,  xx, 352, 256, 3 * N, q,     0, 0);
    lin(stream, hbuf, 352, 304, 3, LN1[1][2], 16,  16,  xx, 352, 304, 3 * N, q,     0, 0);
    hipMemsetAsync(sout, 0, (size_t)N * 640 * 4, stream);
    lin(stream, hbuf, 352, 0,   1, SI[1][0], 256, 256, sout, 640, 0,   N,     inv16, 0, 0);
    lin(stream, hbuf, 352, 0,   1, SI[1][1], 256, 32,  sout, 640, 512, N,     inv16, 0, 0);
    lin(stream, hbuf, 352, 256, 3, SI[1][2], 16,  16,  sout, 640, 544, 3 * N, q,     0, 0);
    lin(stream, hbuf, 352, 304, 3, SI[1][3], 16,  16,  sout, 640, 592, 3 * N, q,     0, 0);
    tp_aggregate<<<N, tb, 0, stream>>>(xx, 352, esh, att, TPWt[1], 608, agg, 1152, 1);
    lin(stream, agg, 1152, 0,    1, LN2[1][0], 256, 256, sout, 640, 0,   N,     sc272, 1, 0);
    lin(stream, agg, 1152, 1072, 1, LN2[1][1], 16,  256, sout, 640, 0,   N,     sc272, 1, 0);
    lin(stream, agg, 1152, 1136, 1, LN2[1][2], 16,  256, sout, 640, 256, N,     sc16g, 1, 0);
    lin(stream, agg, 1152, 0,    1, LN2[1][3], 256, 32,  sout, 640, 512, N,     sc272, 1, 0);
    lin(stream, agg, 1152, 1072, 1, LN2[1][4], 16,  32,  sout, 640, 512, N,     sc272, 1, 0);
    lin(stream, agg, 1152, 256,  3, LN2[1][5], 256, 16,  sout, 640, 544, 3 * N, sc272, 1, 0);
    lin(stream, agg, 1152, 1024, 3, LN2[1][6], 16,  16,  sout, 640, 544, 3 * N, sc272, 1, 0);
    lin(stream, agg, 1152, 1088, 3, LN2[1][7], 16,  16,  sout, 640, 592, 3 * N, sc16g, 1, 0);
    gate_kernel<<<(N * 608 + 255) / 256, tb, 0, stream>>>(sout, 640, hbuf, 608, 256, N * 608);

    // =================== layer 2: (N,608) -> (N,640) -> gate -> (N,608) ===================
    attention_kernel<<<eb, tb, 0, stream>>>(elemb, zemb, ez, KEYS[2], att);
    lin(stream, hbuf, 608, 0,   1, LN1[2][0], 256, 256, xx, 608, 0,   N,     inv16, 0, 0);
    lin(stream, hbuf, 608, 256, 1, LN1[2][1], 256, 256, xx, 608, 256, N,     inv16, 0, 0);
    lin(stream, hbuf, 608, 512, 3, LN1[2][2], 16,  16,  xx, 608, 512, 3 * N, q,     0, 0);
    lin(stream, hbuf, 608, 560, 3, LN1[2][3], 16,  16,  xx, 608, 560, 3 * N, q,     0, 0);
    hipMemsetAsync(sout, 0, (size_t)N * 640 * 4, stream);
    lin(stream, hbuf, 608, 0,   1, SI[2][0], 256, 256, sout, 640, 0,   N,     inv16, 0, 0);
    lin(stream, hbuf, 608, 256, 1, SI[2][1], 256, 256, sout, 640, 256, N,     inv16, 0, 0);
    lin(stream, hbuf, 608, 0,   1, SI[2][2], 256, 32,  sout, 640, 512, N,     inv16, 0, 0);
    lin(stream, hbuf, 608, 512, 3, SI[2][3], 16,  16,  sout, 640, 544, 3 * N, q,     0, 0);
    lin(stream, hbuf, 608, 560, 3, SI[2][4], 16,  16,  sout, 640, 592, 3 * N, q,     0, 0);
    tp_aggregate<<<N, tb, 0, stream>>>(xx, 608, esh, att, TPWt[2], 1120, agg, 2176, 2);
    lin(stream, agg, 2176, 0,    1, LN2[2][0], 256, 256, sout, 640, 0,   N,     sc272, 1, 0);
    lin(stream, agg, 2176, 2096, 1, LN2[2][1], 16,  256, sout, 640, 0,   N,     sc272, 1, 0);
    lin(stream, agg, 2176, 1024, 1, LN2[2][2], 256, 256, sout, 640, 256, N,     sc272, 1, 0);
    lin(stream, agg, 2176, 2160, 1, LN2[2][3], 16,  256, sout, 640, 256, N,     sc272, 1, 0);
    lin(stream, agg, 2176, 0,    1, LN2[2][4], 256, 32,  sout, 640, 512, N,     sc272, 1, 0);
    lin(stream, agg, 2176, 2096, 1, LN2[2][5], 16,  32,  sout, 640, 512, N,     sc272, 1, 0);
    lin(stream, agg, 2176, 256,  3, LN2[2][6], 256, 16,  sout, 640, 544, 3 * N, sc272, 1, 0);
    lin(stream, agg, 2176, 2048, 3, LN2[2][7], 16,  16,  sout, 640, 544, 3 * N, sc272, 1, 0);
    lin(stream, agg, 2176, 1280, 3, LN2[2][8], 256, 16,  sout, 640, 592, 3 * N, sc272, 1, 0);
    lin(stream, agg, 2176, 2112, 3, LN2[2][9], 16,  16,  sout, 640, 592, 3 * N, sc272, 1, 0);
    gate_kernel<<<(N * 608 + 255) / 256, tb, 0, stream>>>(sout, 640, hbuf, 608, 256, N * 608);

    // =================== layer 3 (final): (N,608) -> (N,2) ===================
    attention_kernel<<<eb, tb, 0, stream>>>(elemb, zemb, ez, KEYS[3], att);
    lin(stream, hbuf, 608, 0,   1, LN1[3][0], 256, 256, xx, 608, 0,   N,     inv16, 0, 0);
    lin(stream, hbuf, 608, 256, 1, LN1[3][1], 256, 256, xx, 608, 256, N,     inv16, 0, 0);
    lin(stream, hbuf, 608, 512, 3, LN1[3][2], 16,  16,  xx, 608, 512, 3 * N, q,     0, 0);
    lin(stream, hbuf, 608, 560, 3, LN1[3][3], 16,  16,  xx, 608, 560, 3 * N, q,     0, 0);
    hipMemsetAsync(sout, 0, (size_t)N * 2 * 4, stream);
    lin(stream, hbuf, 608, 0,   1, SI[3][0], 256, 1, sout, 2, 0, N, inv16, 0, 0);
    lin(stream, hbuf, 608, 256, 1, SI[3][1], 256, 1, sout, 2, 1, N, inv16, 0, 0);
    tp_aggregate<<<N, tb, 0, stream>>>(xx, 608, esh, att, TPWt[3], 544, agg, 544, 3);
    lin(stream, agg, 544, 0,   1, LN2[3][0], 256, 1, sout, 2, 0, N, sc272, 1, 0);
    lin(stream, agg, 544, 512, 1, LN2[3][1], 16,  1, sout, 2, 0, N, sc272, 1, 0);
    lin(stream, agg, 544, 256, 1, LN2[3][2], 256, 1, sout, 2, 1, N, sc272, 1, 0);
    lin(stream, agg, 544, 528, 1, LN2[3][3], 16,  1, sout, 2, 1, N, sc272, 1, 0);

    readout_kernel<<<(BMOL + 255) / 256, tb, 0, stream>>>(sout, (float*)d_out);
}
